// TGAT_78606491452338
// MI455X (gfx1250) — matrix operand order repaired and checked by reading, not compiled, not run on hardware
//
#include <hip/hip_runtime.h>
#include <hip/hip_bf16.h>
#include <stdint.h>
#include <stddef.h>

// ---------------- problem constants (match reference) ----------------
#define DD   128
#define TT   128
#define EE   128
#define HH   2
#define DHH  64
#define KN   10
#define NB   1024
#define N0C  (3 * NB)          // 3072
#define N1C  (N0C * KN)        // 30720
#define NUQ  (N1C + N0C)       // 33792

typedef __bf16 bf16;
typedef __attribute__((ext_vector_type(16))) __bf16 v16bf;
typedef __attribute__((ext_vector_type(8)))  float  v8f;
typedef int v4i __attribute__((vector_size(16)));   // matches builtin param type

#define GLOBAL_AS __attribute__((address_space(1)))
#define LDS_AS    __attribute__((address_space(3)))

#if __has_builtin(__builtin_amdgcn_global_load_async_to_lds_b128) && \
    __has_builtin(__builtin_amdgcn_s_wait_asynccnt)
#define HAVE_ASYNC_LDS 1
#else
#define HAVE_ASYNC_LDS 0
#endif

struct alignas(16) B16c { unsigned x[4]; };

// ---------------- input index map ----------------
// params pytree flattened in sorted-key order, then remaining inputs in dict order.
enum {
  IN_W10 = 0, IN_W11, IN_W20, IN_W21, IN_Wk0, IN_Wk1, IN_Wo0, IN_Wo1,
  IN_Wq0, IN_Wq1, IN_Wv0, IN_Wv1, IN_b10, IN_b11, IN_b20, IN_b21,
  IN_bk0, IN_bk1, IN_bo0, IN_bo1, IN_bq0, IN_bq1, IN_bv0, IN_bv1,
  IN_lp_b1, IN_lp_b2, IN_lp_w1, IN_lp_w2, IN_time_b, IN_time_w,
  IN_static_node_feats, IN_nbr_feats_h0, IN_nbr_feats_h1,
  IN_times_h0, IN_times_h1, IN_nbr_times_h0, IN_nbr_times_h1,
  IN_nids_h0, IN_nids_h1, IN_nids_local_h1, IN_nids_local_h0,
  IN_nbrs_h0, IN_nbrs_h1, IN_nbr_local_h0, IN_nbr_mask_h0, IN_nbr_mask_h1,
  IN_src_local, IN_dst_local, IN_neg_local
};

// ---------------- weight convert: W[Kd,128] f32 -> Wt[128,Kd] bf16 ----------------
__global__ void wconv_kernel(const float* __restrict__ W, bf16* __restrict__ Wt, int Kd) {
  int n = blockIdx.x;                      // 0..127 output column
  for (int k = threadIdx.x; k < Kd; k += blockDim.x)
    Wt[(size_t)n * Kd + k] = (bf16)W[(size_t)k * DD + n];
}

// ---------------- build Q input: [nf_gather | time2vec(0)] as bf16 [N,256] ----------------
__global__ void build_q_kernel(const float* __restrict__ sf, const int* __restrict__ nids,
                               const float* __restrict__ tb, bf16* __restrict__ qin, int N) {
  int i = blockIdx.x;
  int t = threadIdx.x;                     // 128 threads
  if (i >= N) return;
  int nid = nids[i];
  qin[(size_t)i * 256 + t]       = (bf16)sf[(size_t)nid * DD + t];
  qin[(size_t)i * 256 + 128 + t] = (bf16)__cosf(tb[t]);   // cos(0*w + b)
}

// ---------------- build KV input: [nbr_feat | edge | time2vec(dt)] bf16 [N*K,384] ------
// nbr_local/zf == nullptr for hop1 (static features only).
__global__ void build_kv_kernel(const float* __restrict__ sf,
                                const int* __restrict__ nbrs,
                                const int* __restrict__ nbr_local,
                                const float* __restrict__ zf,
                                const unsigned char* __restrict__ mask,
                                const float* __restrict__ edge,
                                const float* __restrict__ times,
                                const float* __restrict__ ntimes,
                                const float* __restrict__ tw, const float* __restrict__ tb,
                                bf16* __restrict__ kv, int NK) {
  int row = blockIdx.x;                    // node*K + k
  int t = threadIdx.x;                     // 128 threads
  if (row >= NK) return;
  int node = row / KN;
  bool m = mask[row] != 0;
  float f;
  if (nbr_local != nullptr && m) f = zf[(size_t)nbr_local[row] * DD + t];
  else                           f = sf[(size_t)nbrs[row] * DD + t];
  size_t base = (size_t)row * 384;
  kv[base + t]       = (bf16)f;
  kv[base + 128 + t] = (bf16)edge[(size_t)row * EE + t];
  float dt = m ? (times[node] - ntimes[row]) : 0.0f;
  kv[base + 256 + t] = (bf16)__cosf(dt * tw[t] + tb[t]);
}

// ---------------- shared helpers for WMMA GEMM kernels ----------------
// Stage 64 x KD bf16 A tile into LDS with async global->LDS (ASYNCcnt tracked).
template <int KD>
__device__ inline void stage_A_tile(const bf16* __restrict__ A, bf16* sA,
                                    int row0, int M, int tid) {
  constexpr int CPR   = KD / 8;            // 16B chunks per row
  constexpr int TOTAL = 64 * CPR;
  for (int idx = tid; idx < TOTAL; idx += 256) {
    int r = idx / CPR;
    int c = idx - r * CPR;
    int gr = row0 + r; if (gr >= M) gr = M - 1;
    const bf16* g = A + (size_t)gr * KD + c * 8;
    bf16* l = sA + r * KD + c * 8;
#if HAVE_ASYNC_LDS
    __builtin_amdgcn_global_load_async_to_lds_b128((GLOBAL_AS v4i*)g, (LDS_AS v4i*)l, 0, 0);
#else
    *(B16c*)l = *(const B16c*)g;
#endif
  }
#if HAVE_ASYNC_LDS
  __builtin_amdgcn_s_wait_asynccnt(0);
#endif
  __syncthreads();
}

// B fragment: lanes 0-15 col N=lane16 K={k0..k0+7, k0+16..k0+23}; lanes 16-31 K={k0+8..k0+15, k0+24..k0+31}
__device__ inline v16bf load_b_frag(const bf16* __restrict__ wcol, int k0, int hi) {
  v16bf b;
  int kb = k0 + (hi ? 8 : 0);
#pragma unroll
  for (int j = 0; j < 8; ++j) {
    b[j]     = wcol[kb + j];
    b[8 + j] = wcol[kb + 16 + j];
  }
  return b;
}

// A fragment from LDS: lanes 0-15 row M=lane16, K={ka..ka+7, ka+16..ka+23}
template <int KD>
__device__ inline v16bf load_a_frag(const bf16* sA, int row, int ka) {
  const bf16* ar = sA + (size_t)row * KD;
  v16bf a;
#pragma unroll
  for (int j = 0; j < 8; ++j) {
    a[j]     = ar[ka + j];
    a[8 + j] = ar[ka + 16 + j];
  }
  return a;
}

// ---------------- WMMA GEMM: C[M,128] = act(A[M,KD] @ W[KD,128] + bias) ----------------
// Block = 256 threads = 8 waves; macro-tile = 64 rows x 128 cols.
template <int KD, int ACT, int OUT_BF16>
__global__ void gemm_kernel(const bf16* __restrict__ A, const bf16* __restrict__ Wt,
                            const float* __restrict__ bias,
                            float* __restrict__ Cf, bf16* __restrict__ Cb,
                            int M, int ldc, const int* __restrict__ oidx) {
  extern __shared__ char smem_raw[];
  bf16* sA = (bf16*)smem_raw;              // [64][KD]
  const int tid  = threadIdx.x;
  const int row0 = blockIdx.x * 64;

  stage_A_tile<KD>(A, sA, row0, M, tid);

  const int wave = tid >> 5, lane = tid & 31, lane16 = lane & 15, hi = lane >> 4;
  const int n0 = wave * 16;
  const bf16* __restrict__ wcol = Wt + (size_t)(n0 + lane16) * KD;

  v8f acc[4];
#pragma unroll
  for (int rt = 0; rt < 4; ++rt) acc[rt] = v8f{0.f,0.f,0.f,0.f,0.f,0.f,0.f,0.f};

#pragma unroll
  for (int k0 = 0; k0 < KD; k0 += 32) {
    v16bf bfr = load_b_frag(wcol, k0, hi);
    int ka = k0 + (hi ? 8 : 0);
    v16bf af[4];
#pragma unroll
    for (int rt = 0; rt < 4; ++rt) af[rt] = load_a_frag<KD>(sA, rt * 16 + lane16, ka);
#pragma unroll
    for (int rt = 0; rt < 4; ++rt)
      acc[rt] = __builtin_amdgcn_wmma_f32_16x16x32_bf16(false, af[rt], false, bfr,
                                                        (short)0, acc[rt], false, false);
  }

  const int n = n0 + lane16;
  const float bval = bias ? bias[n] : 0.0f;
#pragma unroll
  for (int rt = 0; rt < 4; ++rt) {
    int mbase = row0 + rt * 16 + hi * 8;
#pragma unroll
    for (int r = 0; r < 8; ++r) {
      int m = mbase + r;
      if (m < M) {
        float v = acc[rt][r] + bval;
        if (ACT) v = v > 0.0f ? v : 0.0f;
        int mo = oidx ? oidx[m] : m;
        if (OUT_BF16) Cb[(size_t)mo * ldc + n] = (bf16)v;
        else          Cf[(size_t)mo * ldc + n] = v;
      }
    }
  }
}

// ---------------- fused dual-output WMMA GEMM (K and V projections) ----------------
// Same tiling as gemm_kernel but computes two outputs sharing the A operand:
// each A fragment feeds 2 WMMAs. Halves global A traffic for the dominant GEMMs.
template <int KD>
__global__ void gemm2_kernel(const bf16* __restrict__ A,
                             const bf16* __restrict__ WtK, const bf16* __restrict__ WtV,
                             const float* __restrict__ biasK, const float* __restrict__ biasV,
                             bf16* __restrict__ Ck, bf16* __restrict__ Cv, int M) {
  extern __shared__ char smem_raw[];
  bf16* sA = (bf16*)smem_raw;              // [64][KD]
  const int tid  = threadIdx.x;
  const int row0 = blockIdx.x * 64;

  stage_A_tile<KD>(A, sA, row0, M, tid);

  const int wave = tid >> 5, lane = tid & 31, lane16 = lane & 15, hi = lane >> 4;
  const int n0 = wave * 16;
  const bf16* __restrict__ wcolK = WtK + (size_t)(n0 + lane16) * KD;
  const bf16* __restrict__ wcolV = WtV + (size_t)(n0 + lane16) * KD;

  v8f acck[4], accv[4];
#pragma unroll
  for (int rt = 0; rt < 4; ++rt) {
    acck[rt] = v8f{0.f,0.f,0.f,0.f,0.f,0.f,0.f,0.f};
    accv[rt] = v8f{0.f,0.f,0.f,0.f,0.f,0.f,0.f,0.f};
  }

#pragma unroll
  for (int k0 = 0; k0 < KD; k0 += 32) {
    v16bf bk = load_b_frag(wcolK, k0, hi);
    v16bf bv = load_b_frag(wcolV, k0, hi);
    int ka = k0 + (hi ? 8 : 0);
    v16bf af[4];
#pragma unroll
    for (int rt = 0; rt < 4; ++rt) af[rt] = load_a_frag<KD>(sA, rt * 16 + lane16, ka);
#pragma unroll
    for (int rt = 0; rt < 4; ++rt) {
      acck[rt] = __builtin_amdgcn_wmma_f32_16x16x32_bf16(false, af[rt], false, bk,
                                                         (short)0, acck[rt], false, false);
      accv[rt] = __builtin_amdgcn_wmma_f32_16x16x32_bf16(false, af[rt], false, bv,
                                                         (short)0, accv[rt], false, false);
    }
  }

  const int n = n0 + lane16;
  const float bvk = biasK[n], bvv = biasV[n];
#pragma unroll
  for (int rt = 0; rt < 4; ++rt) {
    int mbase = row0 + rt * 16 + hi * 8;
#pragma unroll
    for (int r = 0; r < 8; ++r) {
      int m = mbase + r;
      if (m < M) {
        Ck[(size_t)m * DD + n] = (bf16)(acck[rt][r] + bvk);
        Cv[(size_t)m * DD + n] = (bf16)(accv[rt][r] + bvv);
      }
    }
  }
}

// ---------------- attention core: one wave per (node, head) ----------------
__global__ void attention_kernel(const float* __restrict__ q,
                                 const bf16* __restrict__ kk, const bf16* __restrict__ vv,
                                 const unsigned char* __restrict__ mask,
                                 bf16* __restrict__ out, int ldo, int N) {
  int wid = blockIdx.x * (blockDim.x >> 5) + (threadIdx.x >> 5);
  if (wid >= N * HH) return;
  int node = wid >> 1, head = wid & 1;
  int lane = threadIdx.x & 31;
  int dbase = head * DHH + lane * 2;       // each lane owns 2 of the 64 dims

  float q0 = q[(size_t)node * DD + dbase];
  float q1 = q[(size_t)node * DD + dbase + 1];

  float sc[KN];
  float smax = -1e30f;
#pragma unroll
  for (int k = 0; k < KN; ++k) {
    const bf16* kp = kk + ((size_t)node * KN + k) * DD + dbase;
    float s = q0 * (float)kp[0] + q1 * (float)kp[1];
#pragma unroll
    for (int off = 16; off; off >>= 1) s += __shfl_xor(s, off, 32);
    s *= 0.125f;                            // 1/sqrt(64)
    bool valid = mask[(size_t)node * KN + k] != 0;
    s = valid ? s : -1e9f;
    sc[k] = s;
    smax = fmaxf(smax, s);
  }
  bool anyvalid = smax > -1e8f;
  float denom = 0.0f;
#pragma unroll
  for (int k = 0; k < KN; ++k) { sc[k] = __expf(sc[k] - smax); denom += sc[k]; }
  float inv = anyvalid ? 1.0f / denom : 0.0f;

  float o0 = 0.0f, o1 = 0.0f;
#pragma unroll
  for (int k = 0; k < KN; ++k) {
    float a = sc[k] * inv;
    const bf16* vp = vv + ((size_t)node * KN + k) * DD + dbase;
    o0 += a * (float)vp[0];
    o1 += a * (float)vp[1];
  }
  out[(size_t)node * ldo + dbase]     = (bf16)o0;
  out[(size_t)node * ldo + dbase + 1] = (bf16)o1;
}

// ---------------- copy node features into second half of FFN input ----------------
__global__ void copy_nf_kernel(const bf16* __restrict__ qin, bf16* __restrict__ ff1, int N) {
  int i = blockIdx.x, t = threadIdx.x;
  if (i >= N) return;
  ff1[(size_t)i * 256 + 128 + t] = qin[(size_t)i * 256 + t];
}

// ---------------- link-predict input: [z[a] | z[b]] bf16 ----------------
__global__ void build_lp_kernel(const float* __restrict__ z, const int* __restrict__ a,
                                const int* __restrict__ b, bf16* __restrict__ in, int B) {
  int i = blockIdx.x, t = threadIdx.x;
  if (i >= B) return;
  in[(size_t)i * 256 + t]       = (bf16)z[(size_t)a[i] * DD + t];
  in[(size_t)i * 256 + 128 + t] = (bf16)z[(size_t)b[i] * DD + t];
}

// ---------------- final sigmoid(h @ w2 + b2): one wave per row ----------------
__global__ void lp_final_kernel(const bf16* __restrict__ h, const float* __restrict__ w2,
                                const float* __restrict__ b2, float* __restrict__ out, int B) {
  int wid = blockIdx.x * (blockDim.x >> 5) + (threadIdx.x >> 5);
  if (wid >= B) return;
  int lane = threadIdx.x & 31;
  float s = 0.0f;
  for (int d = lane; d < DD; d += 32) s += (float)h[(size_t)wid * DD + d] * w2[d];
#pragma unroll
  for (int off = 16; off; off >>= 1) s += __shfl_xor(s, off, 32);
  if (lane == 0) out[wid] = 1.0f / (1.0f + __expf(-(s + b2[0])));
}

// ---------------- launch helpers ----------------
static inline size_t align_up(size_t x, size_t a) { return (x + a - 1) / a * a; }

template <int KD, int ACT, int OBF>
static void launch_gemm(const bf16* A, const bf16* Wt, const float* bias,
                        float* Cf, bf16* Cb, int M, int ldc, const int* oidx,
                        hipStream_t s) {
  dim3 grid((M + 63) / 64), block(256);
  size_t sh = (size_t)64 * KD * sizeof(bf16);
  gemm_kernel<KD, ACT, OBF><<<grid, block, sh, s>>>(A, Wt, bias, Cf, Cb, M, ldc, oidx);
}

template <int KD>
static void launch_gemm2(const bf16* A, const bf16* WtK, const bf16* WtV,
                         const float* biasK, const float* biasV,
                         bf16* Ck, bf16* Cv, int M, hipStream_t s) {
  dim3 grid((M + 63) / 64), block(256);
  size_t sh = (size_t)64 * KD * sizeof(bf16);
  gemm2_kernel<KD><<<grid, block, sh, s>>>(A, WtK, WtV, biasK, biasV, Ck, Cv, M);
}

extern "C" void kernel_launch(void* const* d_in, const int* in_sizes, int n_in,
                              void* d_out, int out_size, void* d_ws, size_t ws_size,
                              hipStream_t stream) {
  (void)in_sizes; (void)n_in; (void)out_size; (void)ws_size;

  #define FP(i)  ((const float*)d_in[i])
  #define IP(i)  ((const int*)d_in[i])
  #define MP(i)  ((const unsigned char*)d_in[i])

  // ---- workspace carve-out ----
  char* p = (char*)d_ws;
  auto alloc = [&](size_t bytes) { char* r = p; p += align_up(bytes, 256); return (void*)r; };

  // transposed bf16 weights
  bf16* Wq1t = (bf16*)alloc(256 * DD * sizeof(bf16));
  bf16* Wk1t = (bf16*)alloc(384 * DD * sizeof(bf16));
  bf16* Wv1t = (bf16*)alloc(384 * DD * sizeof(bf16));
  bf16* Wo1t = (bf16*)alloc(128 * DD * sizeof(bf16));
  bf16* W11t = (bf16*)alloc(256 * DD * sizeof(bf16));
  bf16* W21t = (bf16*)alloc(128 * DD * sizeof(bf16));
  bf16* Wq0t = (bf16*)alloc(256 * DD * sizeof(bf16));
  bf16* Wk0t = (bf16*)alloc(384 * DD * sizeof(bf16));
  bf16* Wv0t = (bf16*)alloc(384 * DD * sizeof(bf16));
  bf16* Wo0t = (bf16*)alloc(128 * DD * sizeof(bf16));
  bf16* W10t = (bf16*)alloc(256 * DD * sizeof(bf16));
  bf16* W20t = (bf16*)alloc(128 * DD * sizeof(bf16));
  bf16* LPWt = (bf16*)alloc(256 * DD * sizeof(bf16));

  float* zf   = (float*)alloc((size_t)NUQ * DD * sizeof(float));
  bf16*  Qin  = (bf16*) alloc((size_t)N1C * 256 * sizeof(bf16));       // reused per hop + LP
  bf16*  KVin = (bf16*) alloc((size_t)N1C * KN * 384 * sizeof(bf16));
  float* qf   = (float*)alloc((size_t)N1C * DD * sizeof(float));
  bf16*  kkb  = (bf16*) alloc((size_t)N1C * KN * DD * sizeof(bf16));
  bf16*  vvb  = (bf16*) alloc((size_t)N1C * KN * DD * sizeof(bf16));
  bf16*  attn = (bf16*) alloc((size_t)N1C * DD * sizeof(bf16));
  bf16*  FF1  = (bf16*) alloc((size_t)N1C * 256 * sizeof(bf16));
  bf16*  hb   = (bf16*) alloc((size_t)N1C * DD * sizeof(bf16));

  // ---- convert/transpose all GEMM weights to bf16 ----
  {
    dim3 g(DD), b(256);
    wconv_kernel<<<g, b, 0, stream>>>(FP(IN_Wq1), Wq1t, 256);
    wconv_kernel<<<g, b, 0, stream>>>(FP(IN_Wk1), Wk1t, 384);
    wconv_kernel<<<g, b, 0, stream>>>(FP(IN_Wv1), Wv1t, 384);
    wconv_kernel<<<g, b, 0, stream>>>(FP(IN_Wo1), Wo1t, 128);
    wconv_kernel<<<g, b, 0, stream>>>(FP(IN_W11), W11t, 256);
    wconv_kernel<<<g, b, 0, stream>>>(FP(IN_W21), W21t, 128);
    wconv_kernel<<<g, b, 0, stream>>>(FP(IN_Wq0), Wq0t, 256);
    wconv_kernel<<<g, b, 0, stream>>>(FP(IN_Wk0), Wk0t, 384);
    wconv_kernel<<<g, b, 0, stream>>>(FP(IN_Wv0), Wv0t, 384);
    wconv_kernel<<<g, b, 0, stream>>>(FP(IN_Wo0), Wo0t, 128);
    wconv_kernel<<<g, b, 0, stream>>>(FP(IN_W10), W10t, 256);
    wconv_kernel<<<g, b, 0, stream>>>(FP(IN_W20), W20t, 128);
    wconv_kernel<<<g, b, 0, stream>>>(FP(IN_lp_w1), LPWt, 256);
  }

  const float* tw = FP(IN_time_w);
  const float* tb = FP(IN_time_b);
  const float* sf = FP(IN_static_node_feats);

  // ================= hop 1 =================
  {
    const int N = N1C, NK = N1C * KN;
    build_q_kernel<<<dim3(N), dim3(128), 0, stream>>>(sf, IP(IN_nids_h1), tb, Qin, N);
    build_kv_kernel<<<dim3(NK), dim3(128), 0, stream>>>(
        sf, IP(IN_nbrs_h1), nullptr, nullptr, MP(IN_nbr_mask_h1),
        FP(IN_nbr_feats_h1), FP(IN_times_h1), FP(IN_nbr_times_h1), tw, tb, KVin, NK);

    launch_gemm<256, 0, 0>(Qin, Wq1t, FP(IN_bq1), qf, nullptr, N, DD, nullptr, stream);
    launch_gemm2<384>(KVin, Wk1t, Wv1t, FP(IN_bk1), FP(IN_bv1), kkb, vvb, NK, stream);

    attention_kernel<<<dim3((N * HH + 7) / 8), dim3(256), 0, stream>>>(
        qf, kkb, vvb, MP(IN_nbr_mask_h1), attn, DD, N);

    launch_gemm<128, 0, 1>(attn, Wo1t, FP(IN_bo1), nullptr, FF1, N, 256, nullptr, stream);
    copy_nf_kernel<<<dim3(N), dim3(128), 0, stream>>>(Qin, FF1, N);
    launch_gemm<256, 1, 1>(FF1, W11t, FP(IN_b11), nullptr, hb, N, DD, nullptr, stream);
    launch_gemm<128, 0, 0>(hb, W21t, FP(IN_b21), zf, nullptr, N, DD,
                           IP(IN_nids_local_h1), stream);
  }

  // ================= hop 0 =================
  {
    const int N = N0C, NK = N0C * KN;
    build_q_kernel<<<dim3(N), dim3(128), 0, stream>>>(sf, IP(IN_nids_h0), tb, Qin, N);
    build_kv_kernel<<<dim3(NK), dim3(128), 0, stream>>>(
        sf, IP(IN_nbrs_h0), IP(IN_nbr_local_h0), zf, MP(IN_nbr_mask_h0),
        FP(IN_nbr_feats_h0), FP(IN_times_h0), FP(IN_nbr_times_h0), tw, tb, KVin, NK);

    launch_gemm<256, 0, 0>(Qin, Wq0t, FP(IN_bq0), qf, nullptr, N, DD, nullptr, stream);
    launch_gemm2<384>(KVin, Wk0t, Wv0t, FP(IN_bk0), FP(IN_bv0), kkb, vvb, NK, stream);

    attention_kernel<<<dim3((N * HH + 7) / 8), dim3(256), 0, stream>>>(
        qf, kkb, vvb, MP(IN_nbr_mask_h0), attn, DD, N);

    launch_gemm<128, 0, 1>(attn, Wo0t, FP(IN_bo0), nullptr, FF1, N, 256, nullptr, stream);
    copy_nf_kernel<<<dim3(N), dim3(128), 0, stream>>>(Qin, FF1, N);
    launch_gemm<256, 1, 1>(FF1, W10t, FP(IN_b10), nullptr, hb, N, DD, nullptr, stream);
    launch_gemm<128, 0, 0>(hb, W20t, FP(IN_b20), zf, nullptr, N, DD,
                           IP(IN_nids_local_h0), stream);
  }

  // ================= link prediction =================
  {
    float* out = (float*)d_out;                 // [2048] = pos(1024) | neg(1024)
    // positive pairs
    build_lp_kernel<<<dim3(NB), dim3(128), 0, stream>>>(
        zf, IP(IN_src_local), IP(IN_dst_local), Qin, NB);
    launch_gemm<256, 1, 1>(Qin, LPWt, FP(IN_lp_b1), nullptr, hb, NB, DD, nullptr, stream);
    lp_final_kernel<<<dim3((NB + 7) / 8), dim3(256), 0, stream>>>(
        hb, FP(IN_lp_w2), FP(IN_lp_b2), out, NB);
    // negative pairs
    build_lp_kernel<<<dim3(NB), dim3(128), 0, stream>>>(
        zf, IP(IN_src_local), IP(IN_neg_local), Qin, NB);
    launch_gemm<256, 1, 1>(Qin, LPWt, FP(IN_lp_b1), nullptr, hb, NB, DD, nullptr, stream);
    lp_final_kernel<<<dim3((NB + 7) / 8), dim3(256), 0, stream>>>(
        hb, FP(IN_lp_w2), FP(IN_lp_b2), out + NB, NB);
  }

  #undef FP
  #undef IP
  #undef MP
}
